// PMBNNCell_45569603010671
// MI455X (gfx1250) — matrix operand order repaired and checked by reading, not compiled, not run on hardware
//
#include <hip/hip_runtime.h>
#include <hip/hip_bf16.h>

// ---------- problem constants (from reference) ----------
#define BATCH   4096
#define DIN     784
#define KPAD    800          // 784 padded to multiple of 32 for K-loop
#define DH      128
#define DLAT    8
#define CH      32
#define DOUTN   10
#define NC      16
#define TSTEPS  5
#define PM_STEPS 4
#define DT_     0.15f
#define BETA_   1.0f
#define EPS_    1e-4f
#define KE_     0.8f
#define KI_     1.0f
#define INV2SI  (1.0f / 2.88f)   // 1/(2*1.2^2); note 1/(2*0.6^2) = 4x this

// ---------- vector types ----------
typedef __attribute__((ext_vector_type(16))) __bf16 v16bf;
typedef __attribute__((ext_vector_type(8)))  float  v8f;
typedef __attribute__((ext_vector_type(4)))  unsigned u32x4;
typedef __attribute__((ext_vector_type(4)))  float  f32x4;

union BFreg { v16bf v; u32x4 q[2]; unsigned short s[16]; };

__device__ __forceinline__ unsigned short bf16bits(float f) {
    union { float f; unsigned u; } x; x.f = f;
    unsigned r = x.u + 0x7FFFu + ((x.u >> 16) & 1u);   // round-to-nearest-even
    return (unsigned short)(r >> 16);
}

// ---------- workspace layout (bytes) ----------
#define OFF_XBF  0u                                    // [4096][800] bf16
#define OFF_W1T  (OFF_XBF + BATCH*KPAD*2u)             // [128][800]  bf16 (W1 transposed)
#define OFF_H1   (OFF_W1T + DH*KPAD*2u)                // [4096][128] f32
#define OFF_Z    (OFF_H1  + BATCH*DH*4u)               // [4096][8]   f32
#define OFF_H0   (OFF_Z   + BATCH*DLAT*4u)             // [4096][32]  f32
#define OFF_HM   (OFF_H0  + BATCH*CH*4u)               // [4096][32]  f32 (h_mid)
#define OFF_HB   (OFF_HM  + BATCH*CH*4u)               // [128][2][32 lanes][16 bf16] staged B-operand

// =====================================================================
// K0: convert x -> bf16, zero-padded to KPAD
__global__ void __launch_bounds__(256) k_cvt_x(const float* __restrict__ x,
                                               unsigned short* __restrict__ xbf) {
    int idx = blockIdx.x * 256 + threadIdx.x;
    if (idx >= BATCH * KPAD) return;
    int r = idx / KPAD, k = idx % KPAD;
    xbf[idx] = (k < DIN) ? bf16bits(x[r * DIN + k]) : (unsigned short)0;
}

// K1: convert W1 -> W1^T bf16 [128][800]
__global__ void __launch_bounds__(256) k_cvt_w1t(const float* __restrict__ W1,
                                                 unsigned short* __restrict__ w1t) {
    int idx = blockIdx.x * 256 + threadIdx.x;
    if (idx >= DH * KPAD) return;
    int n = idx / KPAD, k = idx % KPAD;
    w1t[idx] = (k < DIN) ? bf16bits(W1[k * DH + n]) : (unsigned short)0;
}

// =====================================================================
// K2: H1 = tanh(x @ W1 + b1)  via v_wmma_f32_16x16x32_bf16
// grid 256 blocks (16-row slabs), 8 waves/block (16-col tiles each)
__global__ void __launch_bounds__(256) k_encoder(const unsigned short* __restrict__ xbf,
                                                 const unsigned short* __restrict__ w1t,
                                                 const float* __restrict__ b1,
                                                 float* __restrict__ H1) {
    const int tid  = threadIdx.x;
    const int wave = tid >> 5;
    const int lane = tid & 31;
    const int il   = lane & 15;
    const int rowbase = blockIdx.x * 16;
    const int colbase = wave * 16;

    // A-operand addressing (16-bit A layout, ISA 7.12.2):
    // lane<16: M=lane,   K in {0..7, 16..23};  lane>=16: M=lane-16, K in {8..15, 24..31}
    const int abase = (lane < 16) ? 0 : 8;
    const unsigned short* arow = xbf + (rowbase + il) * KPAD + abase;
    // B-operand: lane holds column N=lane%16; lanes<16 K in {0..7, 16..23}, lanes>=16 K in {8..15, 24..31}
    const int bbase = (lane < 16) ? 0 : 8;
    const unsigned short* bcol = w1t + (colbase + il) * KPAD + bbase;

    v8f acc = {};
    for (int kt = 0; kt < KPAD / 32; ++kt) {
        const unsigned short* pa = arow + kt * 32;
        const unsigned short* pb = bcol + kt * 32;
        // prefetch a few K-tiles ahead of the streamed operands into the near
        // cache (locality=3 -> WGP-scope global_prefetch_b8)
        __builtin_prefetch(pa + 4 * 32, 0, 3);
        __builtin_prefetch(pb + 4 * 32, 0, 3);
        BFreg A, Bm;
        A.q[0]  = *(const u32x4*)(pa);        // K = base+0..7
        A.q[1]  = *(const u32x4*)(pa + 16);   // K = base+16..23
        Bm.q[0] = *(const u32x4*)(pb);        // K = bbase+0..7
        Bm.q[1] = *(const u32x4*)(pb + 16);   // K = bbase+16..23
        acc = __builtin_amdgcn_wmma_f32_16x16x32_bf16(false, A.v, false, Bm.v,
                                                      (short)0, acc, false, false);
    }
    // C layout: VGPR r -> M = r + (lane<16?0:8), N = lane%16
    const int mofs = (lane < 16) ? 0 : 8;
    const int col  = colbase + il;
    const float bb = b1[col];
#pragma unroll
    for (int r = 0; r < 8; ++r) {
        int row = rowbase + r + mofs;
        H1[row * DH + col] = tanhf(acc[r] + bb);
    }
}

// =====================================================================
// K3: z = H1 @ W2 + b2   (4096 x 128 x 8, tiny)
__global__ void __launch_bounds__(256) k_zinit(const float* __restrict__ H1,
                                               const float* __restrict__ W2,
                                               const float* __restrict__ b2,
                                               float* __restrict__ z) {
    int idx = blockIdx.x * 256 + threadIdx.x;
    if (idx >= BATCH * DLAT) return;
    int row = idx >> 3, d = idx & 7;
    float acc = b2[d];
    const float* hr = H1 + row * DH;
    for (int k = 0; k < DH; ++k) acc += hr[k] * W2[k * DLAT + d];
    z[idx] = acc;
}

// =====================================================================
// K5: PM field — 4 explicit-Euler steps per row
__global__ void __launch_bounds__(256) k_pm(float* __restrict__ z,
                                            const float* __restrict__ centers,
                                            const float* __restrict__ mus) {
    __shared__ float cen[NC * DLAT];
    __shared__ float cmu[NC];
    int tid = threadIdx.x;
    if (tid < NC * DLAT) cen[tid] = centers[tid];
    if (tid < NC)        cmu[tid] = mus[tid];
    __syncthreads();

    int row = blockIdx.x * 256 + tid;
    float zr[DLAT];
#pragma unroll
    for (int d = 0; d < DLAT; ++d) zr[d] = z[row * DLAT + d];

    for (int s = 0; s < PM_STEPS; ++s) {
        float n = 1.0f;
        float g[DLAT];
#pragma unroll
        for (int d = 0; d < DLAT; ++d) g[d] = 0.0f;
        for (int c = 0; c < NC; ++c) {
            float dv[DLAT], r2 = 0.0f;
#pragma unroll
            for (int d = 0; d < DLAT; ++d) {
                dv[d] = zr[d] - cen[c * DLAT + d];
                r2 += dv[d] * dv[d];
            }
            float r2e  = r2 + EPS_;
            float rr   = sqrtf(r2e);
            float muor = cmu[c] / rr;     // mu / r
            n += muor;
            float coef = muor / r2e;      // mu / r^3
#pragma unroll
            for (int d = 0; d < DLAT; ++d) g[d] -= coef * dv[d];
        }
        float inv_n = (DT_ * BETA_) / n;
#pragma unroll
        for (int d = 0; d < DLAT; ++d)
            zr[d] = fminf(fmaxf(zr[d] + inv_n * g[d], -3.0f), 3.0f);
    }
#pragma unroll
    for (int d = 0; d < DLAT; ++d) z[row * DLAT + d] = zr[d];
}

// =====================================================================
// K6: h_mid = 0.9 h + 0.1 tanh(z@Wp+bp); also stage h_mid as bf16 in the
// contiguous-K B-operand layout k_lateral uses: tile jt (32 rows), ctile (16 cols), lane = c%16 + 16*(jl>=16),
// slot s = jl%16 (lanes<16 hold K=0..15, lanes>=16 hold K=16..31).
__global__ void __launch_bounds__(256) k_proj(const float* __restrict__ z,
                                              const float* __restrict__ Wp,
                                              const float* __restrict__ bp,
                                              const float* __restrict__ h_in,
                                              float* __restrict__ h_mid,
                                              unsigned short* __restrict__ hB) {
    int idx = blockIdx.x * 256 + threadIdx.x;
    if (idx >= BATCH * CH) return;
    int row = idx >> 5, c = idx & 31;
    float s = bp[c];
    const float* zr = z + row * DLAT;
#pragma unroll
    for (int d = 0; d < DLAT; ++d) s += zr[d] * Wp[d * CH + c];
    float hm = 0.9f * h_in[idx] + 0.1f * tanhf(s);
    h_mid[idx] = hm;

    int jt = row >> 5, jl = row & 31;
    int lane = (c & 15) | ((jl >> 4) << 4);
    int slot = jl & 15;
    int ct   = c >> 4;
    hB[(((jt * 2 + ct) * 32) + lane) * 16 + slot] = bf16bits(hm);
}

// =====================================================================
// K7: fused lateral kernel: h_out = h_mid + 0.05 * K(z) @ h_mid
// K tile computed per-lane directly into WMMA A-operand registers; whole z
// plus per-row |z|^2 staged in LDS. dist2 via norm trick; single v_exp_f32
// per pair since exp(-d2/0.72) = exp(-d2/2.88)^4.
__global__ void __launch_bounds__(256) k_lateral(const float* __restrict__ z,
                                                 const float* __restrict__ h_mid,
                                                 const unsigned short* __restrict__ hB,
                                                 float* __restrict__ h_out) {
    extern __shared__ float smem[];
    float* zsh  = smem;                    // [4096][8]  z copy      (32768 f)
    float* zsq  = smem + BATCH * DLAT;     // [4096]     |z_j|^2     ( 4096 f)
    float* accs = zsq  + BATCH;            // [16][32]   reduction   (  512 f)

    const int tid  = threadIdx.x;
    // stage z (128 KB) cooperatively
    for (int i = tid; i < BATCH * DLAT / 4; i += 256)
        ((f32x4*)zsh)[i] = ((const f32x4*)z)[i];
    // per-row squared norms (read from global: independent of LDS staging)
    for (int r = tid; r < BATCH; r += 256) {
        const float* zr = z + r * DLAT;
        float sq = 0.0f;
#pragma unroll
        for (int d = 0; d < DLAT; ++d) sq += zr[d] * zr[d];
        zsq[r] = sq;
    }
    accs[tid]       = 0.0f;
    accs[tid + 256] = 0.0f;
    __syncthreads();

    const int wave = tid >> 5;
    const int lane = tid & 31;
    const int il   = lane & 15;
    const int irow = blockIdx.x * 16 + il;

    float zi[DLAT];
#pragma unroll
    for (int d = 0; d < DLAT; ++d) zi[d] = zsh[irow * DLAT + d];
    const float sqi = zsq[irow];

    const int abase = (lane < 16) ? 0 : 16;   // A-layout K-base for this lane
    v8f acc0 = {}, acc1 = {};

    for (int jt = wave; jt < BATCH / 32; jt += 8) {
        // prefetch next staged-B tile for this wave into the near cache
        __builtin_prefetch(hB + (((jt + 8) * 2) * 32) * 16, 0, 3);
        // build A = K-kernel tile [16 x 32] in bf16, contiguous K order (k = 16*(lane/16) + s)
        BFreg A;
#pragma unroll
        for (int s = 0; s < 16; ++s) {
            int jl = abase + s;
            int j  = jt * 32 + jl;
            const float* zj = zsh + j * DLAT;
            float dot = 0.0f;
#pragma unroll
            for (int d = 0; d < DLAT; ++d) dot = fmaf(zi[d], zj[d], dot);
            float d2 = fmaxf(sqi + zsq[j] - 2.0f * dot, 0.0f);
            float e  = __expf(-d2 * INV2SI);     // exp(-d2/2.88)
            float e2 = e * e;
            float kv = KE_ * (e2 * e2) - KI_ * e; // 0.8*exp(-d2/0.72) - exp(-d2/2.88)
            A.s[s] = bf16bits(kv);
        }
        // B operands from staged h_mid (bf16, exact B layout): 32 B / lane / ctile
        const u32x4* pb0 = (const u32x4*)(hB + (((jt * 2 + 0) * 32) + lane) * 16);
        const u32x4* pb1 = (const u32x4*)(hB + (((jt * 2 + 1) * 32) + lane) * 16);
        BFreg B0, B1;
        B0.q[0] = pb0[0]; B0.q[1] = pb0[1];
        B1.q[0] = pb1[0]; B1.q[1] = pb1[1];
        acc0 = __builtin_amdgcn_wmma_f32_16x16x32_bf16(false, A.v, false, B0.v,
                                                       (short)0, acc0, false, false);
        acc1 = __builtin_amdgcn_wmma_f32_16x16x32_bf16(false, A.v, false, B1.v,
                                                       (short)0, acc1, false, false);
    }

    // cross-wave reduction into LDS (C layout: VGPR r -> M = r + (lane<16?0:8), N = lane%16)
    const int mofs = (lane < 16) ? 0 : 8;
#pragma unroll
    for (int r = 0; r < 8; ++r) {
        int m = r + mofs;
        atomicAdd(&accs[m * CH + il],      acc0[r]);
        atomicAdd(&accs[m * CH + 16 + il], acc1[r]);
    }
    __syncthreads();

    // finalize: h_out = h_mid + 0.05 * acc
#pragma unroll
    for (int e = tid; e < 16 * CH; e += 256) {
        int m = e >> 5, c = e & 31;
        int gi = (blockIdx.x * 16 + m) * CH + c;
        h_out[gi] = h_mid[gi] + 0.05f * accs[e];
    }
}

// =====================================================================
// K8: y = h @ Wr + br
__global__ void __launch_bounds__(256) k_readout(const float* __restrict__ h,
                                                 const float* __restrict__ Wr,
                                                 const float* __restrict__ br,
                                                 float* __restrict__ y) {
    int idx = blockIdx.x * 256 + threadIdx.x;
    if (idx >= BATCH * DOUTN) return;
    int row = idx / DOUTN, o = idx % DOUTN;
    float acc = br[o];
    const float* hr = h + row * CH;
#pragma unroll
    for (int c = 0; c < CH; ++c) acc += hr[c] * Wr[c * DOUTN + o];
    y[idx] = acc;
}

// =====================================================================
extern "C" void kernel_launch(void* const* d_in, const int* in_sizes, int n_in,
                              void* d_out, int out_size, void* d_ws, size_t ws_size,
                              hipStream_t stream) {
    (void)in_sizes; (void)n_in; (void)out_size; (void)ws_size;
    const float* x       = (const float*)d_in[0];
    const float* W1      = (const float*)d_in[1];
    const float* b1      = (const float*)d_in[2];
    const float* W2      = (const float*)d_in[3];
    const float* b2      = (const float*)d_in[4];
    const float* centers = (const float*)d_in[5];
    const float* mus     = (const float*)d_in[6];
    const float* Wp      = (const float*)d_in[7];
    const float* bp      = (const float*)d_in[8];
    const float* Wr      = (const float*)d_in[9];
    const float* br      = (const float*)d_in[10];
    // d_in[11] is T (int scalar on device) — reference uses T=5, fixed.

    char* ws = (char*)d_ws;
    unsigned short* xbf = (unsigned short*)(ws + OFF_XBF);
    unsigned short* w1t = (unsigned short*)(ws + OFF_W1T);
    float* H1  = (float*)(ws + OFF_H1);
    float* z   = (float*)(ws + OFF_Z);
    float* h0  = (float*)(ws + OFF_H0);
    float* hm  = (float*)(ws + OFF_HM);
    unsigned short* hB = (unsigned short*)(ws + OFF_HB);
    float* y   = (float*)d_out;

    // encoder
    k_cvt_x  <<<(BATCH * KPAD + 255) / 256, 256, 0, stream>>>(x, xbf);
    k_cvt_w1t<<<(DH * KPAD + 255) / 256,    256, 0, stream>>>(W1, w1t);
    k_encoder<<<BATCH / 16, 256, 0, stream>>>(xbf, w1t, b1, H1);
    k_zinit  <<<(BATCH * DLAT + 255) / 256, 256, 0, stream>>>(H1, W2, b2, z);
    hipMemsetAsync(h0, 0, BATCH * CH * sizeof(float), stream);

    // recurrent iterations
    const unsigned smem_bytes = (BATCH * DLAT + BATCH + 16 * CH) * sizeof(float); // 149504 B
    for (int t = 0; t < TSTEPS; ++t) {
        k_pm     <<<BATCH / 256, 256, 0, stream>>>(z, centers, mus);
        k_proj   <<<(BATCH * CH + 255) / 256, 256, 0, stream>>>(z, Wp, bp, h0, hm, hB);
        k_lateral<<<BATCH / 16, 256, smem_bytes, stream>>>(z, hm, hB, h0);
    }

    // readout
    k_readout<<<(BATCH * DOUTN + 255) / 256, 256, 0, stream>>>(h0, Wr, br, y);
}
